// ContactGraspModelFusion_49082886258879
// MI455X (gfx1250) — matrix operand order repaired and checked by reading, not compiled, not run on hardware
//
#include <hip/hip_runtime.h>
#include <hip/hip_bf16.h>
#include <math.h>

#define INV_BN 0.9999950000374997f
#define NEG_INF -3.4e38f

typedef __bf16 v16bf __attribute__((ext_vector_type(16)));
typedef __bf16 v8bf  __attribute__((ext_vector_type(8)));
typedef float  v8f   __attribute__((ext_vector_type(8)));

__device__ inline __bf16 f2bf(float f) {
  unsigned u = __builtin_bit_cast(unsigned, f);
  unsigned r = u + 0x7FFFu + ((u >> 16) & 1u);   // round-to-nearest-even
  unsigned short h = (unsigned short)(r >> 16);
  return __builtin_bit_cast(__bf16, h);
}
__device__ inline float bf2f(__bf16 h) {
  unsigned x = ((unsigned)__builtin_bit_cast(unsigned short, h)) << 16;
  return __builtin_bit_cast(float, x);
}
__device__ inline v16bf cat8(v8bf lo, v8bf hi) {
  return __builtin_shufflevector(lo, hi, 0,1,2,3,4,5,6,7,8,9,10,11,12,13,14,15);
}
// A-fragment for 16x16x32 bf16 WMMA from a row pointer (stride-aligned, k0 mult 32).
// lane: half = lane>>4. elements i<8 -> k0+8*half+i ; i>=8 -> k0+16+8*half+(i-8)
__device__ inline v16bf load_a_frag(const __bf16* rowk0, int half) {
  const v8bf* p = (const v8bf*)rowk0;
  return cat8(p[half], p[2 + half]);
}
// B-fragment: same k order as A: i<8 -> k0+8*half+i ; i>=8 -> k0+16+8*half+(i-8), column (lane&15)
__device__ inline v16bf load_b_frag(const __bf16* colk0, int half) {
  const v8bf* p = (const v8bf*)colk0;
  return cat8(p[half], p[2 + half]);
}

// ---------------------------------------------------------------------------
// Weight conversion: f32 [Cout,K] -> bf16 [Cout,Kp], zero padded
// ---------------------------------------------------------------------------
__global__ void wcvt_kernel(const float* __restrict__ W, int K, int Kp, int Cout,
                            __bf16* __restrict__ out) {
  long i = (long)blockIdx.x * blockDim.x + threadIdx.x;
  if (i >= (long)Cout * Kp) return;
  int k = (int)(i % Kp);
  long co = i / Kp;
  out[i] = (k < K) ? f2bf(W[co * K + k]) : f2bf(0.f);
}

// ---------------------------------------------------------------------------
// Data movement kernels
// ---------------------------------------------------------------------------
__global__ void transpose_cn_kernel(const float* __restrict__ in, int Ctot, int Npts,
                                    int c0, int Csel, float* __restrict__ out) {
  long i = (long)blockIdx.x * blockDim.x + threadIdx.x;
  if (i >= 4L * Npts * Csel) return;
  int c = (int)(i % Csel);
  long rn = i / Csel;
  int n = (int)(rn % Npts);
  int b = (int)(rn / Npts);
  out[i] = in[((long)b * Ctot + c0 + c) * Npts + n];
}

__global__ void transpose_cn_bf_kernel(const float* __restrict__ in, int Ctot, int Npts,
                                       int c0, int Csel, __bf16* __restrict__ out) {
  long i = (long)blockIdx.x * blockDim.x + threadIdx.x;
  if (i >= 4L * Npts * Csel) return;
  int c = (int)(i % Csel);
  long rn = i / Csel;
  int n = (int)(rn % Npts);
  int b = (int)(rn / Npts);
  out[i] = f2bf(in[((long)b * Ctot + c0 + c) * Npts + n]);
}

__global__ void copy_bf_cols_kernel(const __bf16* __restrict__ src, int Cs,
                                    __bf16* __restrict__ dst, int stride, int off, long rows) {
  long i = (long)blockIdx.x * blockDim.x + threadIdx.x;
  if (i >= rows * Cs) return;
  long r = i / Cs; int c = (int)(i % Cs);
  dst[r * stride + off + c] = src[r * Cs + c];
}

__global__ void copy_f32_bf_cols_kernel(const float* __restrict__ src, int Cs,
                                        __bf16* __restrict__ dst, int stride, int off, long rows) {
  long i = (long)blockIdx.x * blockDim.x + threadIdx.x;
  if (i >= rows * Cs) return;
  long r = i / Cs; int c = (int)(i % Cs);
  dst[r * stride + off + c] = f2bf(src[r * Cs + c]);
}

__global__ void zero_cols_kernel(__bf16* __restrict__ dst, int stride, int start, long rows) {
  int w = stride - start;
  long i = (long)blockIdx.x * blockDim.x + threadIdx.x;
  if (i >= rows * w) return;
  long r = i / w; int c = (int)(i % w);
  dst[r * stride + start + c] = f2bf(0.f);
}

__global__ void broadcast_rows_kernel(const float* __restrict__ src, int C,
                                      __bf16* __restrict__ dst, int stride, int off, int Npts) {
  long i = (long)blockIdx.x * blockDim.x + threadIdx.x;
  if (i >= 4L * Npts * C) return;
  int c = (int)(i % C);
  long rn = i / C;
  int b = (int)(rn / Npts);
  dst[rn * stride + off + c] = f2bf(src[(long)b * C + c]);
}

__global__ void gather_rows_kernel(const float* __restrict__ src, int Nin,
                                   const int* __restrict__ idx, int Q,
                                   float* __restrict__ out) {
  long i = (long)blockIdx.x * blockDim.x + threadIdx.x;
  if (i >= 4L * Q * 3) return;
  int c = (int)(i % 3);
  long rq = i / 3;
  int b = (int)(rq / Q);
  out[i] = src[((long)b * Nin + idx[rq]) * 3 + c];
}

__global__ void max_rows_kernel(const __bf16* __restrict__ in, int R, int C,
                                float* __restrict__ out) {
  long i = (long)blockIdx.x * blockDim.x + threadIdx.x;
  if (i >= 4L * C) return;
  int b = (int)(i / C), c = (int)(i % C);
  const __bf16* p = in + ((long)b * R) * C + c;
  float m = NEG_INF;
  for (int r = 0; r < R; ++r) m = fmaxf(m, bf2f(p[(long)r * C]));
  out[(long)b * C + c] = m;
}

// ---------------------------------------------------------------------------
// Farthest point sampling: one workgroup per batch, dist array in LDS.
// ---------------------------------------------------------------------------
__global__ __launch_bounds__(256)
void fps_kernel(const float* __restrict__ xyz, int N, int npoint, int* __restrict__ fidx) {
  __shared__ float dist[4000];
  __shared__ float rv[256];
  __shared__ int   ri[256];
  __shared__ int   far_s;
  int b = blockIdx.x, tid = threadIdx.x;
  for (int i = tid; i < N; i += 256) dist[i] = 1e10f;
  if (tid == 0) far_s = 0;
  __syncthreads();
  for (int it = 0; it < npoint; ++it) {
    int far = far_s;
    if (tid == 0) fidx[(long)b * npoint + it] = far;
    float cx = xyz[((long)b * N + far) * 3 + 0];
    float cy = xyz[((long)b * N + far) * 3 + 1];
    float cz = xyz[((long)b * N + far) * 3 + 2];
    float bestv = NEG_INF; int besti = 0;
    for (int i = tid; i < N; i += 256) {
      float dx = xyz[((long)b * N + i) * 3 + 0] - cx;
      float dy = xyz[((long)b * N + i) * 3 + 1] - cy;
      float dz = xyz[((long)b * N + i) * 3 + 2] - cz;
      float nd = fminf(dist[i], dx * dx + dy * dy + dz * dz);
      dist[i] = nd;
      if (nd > bestv) { bestv = nd; besti = i; }
    }
    rv[tid] = bestv; ri[tid] = besti;
    __syncthreads();
    for (int s = 128; s > 0; s >>= 1) {
      if (tid < s) {
        if (rv[tid + s] > rv[tid] ||
            (rv[tid + s] == rv[tid] && ri[tid + s] < ri[tid])) {
          rv[tid] = rv[tid + s]; ri[tid] = ri[tid + s];
        }
      }
      __syncthreads();
    }
    if (tid == 0) far_s = ri[0];
    __syncthreads();
  }
}

// ---------------------------------------------------------------------------
// Ball query: one wave per query point, ordered compaction via ballot/popcount.
// ---------------------------------------------------------------------------
__global__ __launch_bounds__(128)
void query_ball_kernel(const float* __restrict__ xyz, int N,
                       const float* __restrict__ newxyz, int Q,
                       float r2, int K, int* __restrict__ gidx) {
  int gw = blockIdx.x * (blockDim.x >> 5) + (threadIdx.x >> 5);
  int lane = threadIdx.x & 31;
  if (gw >= 4 * Q) return;
  int b = gw / Q, q = gw % Q;
  float cx = newxyz[((long)b * Q + q) * 3 + 0];
  float cy = newxyz[((long)b * Q + q) * 3 + 1];
  float cz = newxyz[((long)b * Q + q) * 3 + 2];
  int cnt = 0, first = -1;
  int* out = gidx + ((long)b * Q + q) * K;
  for (int base = 0; base < N && cnt < K; base += 32) {
    int i = base + lane;
    bool in = false;
    if (i < N) {
      float dx = xyz[((long)b * N + i) * 3 + 0] - cx;
      float dy = xyz[((long)b * N + i) * 3 + 1] - cy;
      float dz = xyz[((long)b * N + i) * 3 + 2] - cz;
      in = (dx * dx + dy * dy + dz * dz) <= r2;
    }
    unsigned mask = (unsigned)__ballot(in);
    if (first < 0 && mask) first = base + __builtin_ctz(mask);
    int pos = cnt + __popc(mask & ((1u << lane) - 1u));
    if (in && pos < K) out[pos] = i;
    cnt += __popc(mask);
  }
  if (cnt > K) cnt = K;
  if (first < 0) first = 0;
  for (int j = cnt + lane; j < K; j += 32) out[j] = first;
}

// ---------------------------------------------------------------------------
// 3-NN inverse-distance interpolation, bf16 features -> bf16 concat buffer.
// ---------------------------------------------------------------------------
__global__ void knn3_interp_kernel(const float* __restrict__ xyz1, int N1,
                                   const float* __restrict__ xyz2, int S,
                                   const __bf16* __restrict__ p2, int C,
                                   __bf16* __restrict__ out, int stride, int off) {
  long i = (long)blockIdx.x * blockDim.x + threadIdx.x;
  if (i >= 4L * N1) return;
  int b = (int)(i / N1);
  float px = xyz1[i * 3 + 0], py = xyz1[i * 3 + 1], pz = xyz1[i * 3 + 2];
  float d0 = 1e30f, d1 = 1e30f, d2 = 1e30f;
  int i0 = 0, i1 = 0, i2 = 0;
  for (int s = 0; s < S; ++s) {
    float dx = xyz2[((long)b * S + s) * 3 + 0] - px;
    float dy = xyz2[((long)b * S + s) * 3 + 1] - py;
    float dz = xyz2[((long)b * S + s) * 3 + 2] - pz;
    float d = dx * dx + dy * dy + dz * dz;
    if (d < d0)      { d2 = d1; i2 = i1; d1 = d0; i1 = i0; d0 = d; i0 = s; }
    else if (d < d1) { d2 = d1; i2 = i1; d1 = d;  i1 = s; }
    else if (d < d2) { d2 = d;  i2 = s; }
  }
  float w0 = 1.f / (d0 + 1e-8f), w1 = 1.f / (d1 + 1e-8f), w2 = 1.f / (d2 + 1e-8f);
  float ws = w0 + w1 + w2; w0 /= ws; w1 /= ws; w2 /= ws;
  const __bf16* r0 = p2 + ((long)b * S + i0) * C;
  const __bf16* r1 = p2 + ((long)b * S + i1) * C;
  const __bf16* r2 = p2 + ((long)b * S + i2) * C;
  __bf16* o = out + i * stride + off;
  for (int c = 0; c < C; ++c)
    o[c] = f2bf(w0 * bf2f(r0[c]) + w1 * bf2f(r1[c]) + w2 * bf2f(r2[c]));
}

// ---------------------------------------------------------------------------
// Generic WMMA GEMM, fully bounds-check-free:
//   A: bf16 [R, Kp] (R mult 32, Kp mult 32, zero-padded cols)
//   W: bf16 [Cout, Kp] (pre-converted, zero-padded)
//   Out: bf16 [R, Cout] = act((A*W^T + bias) * scale)
// One wave per 32x32 tile (2x2 WMMA micro-tile).
// ---------------------------------------------------------------------------
__global__ __launch_bounds__(32)
void wmma_gemm_kernel(const __bf16* __restrict__ A, const __bf16* __restrict__ W,
                      const float* __restrict__ bias, __bf16* __restrict__ Out,
                      int Kp, int Cout, float scale, int relu) {
  int lane = threadIdx.x & 31;
  int half = lane >> 4, mr = lane & 15;
  long row0 = (long)blockIdx.x * 32;
  int  col0 = blockIdx.y * 32;
  const __bf16* a0p = A + (row0 + mr) * Kp;
  const __bf16* a1p = A + (row0 + 16 + mr) * Kp;
  const __bf16* b0p = W + (long)(col0 + mr) * Kp;
  const __bf16* b1p = W + (long)(col0 + 16 + mr) * Kp;
  v8f c00, c01, c10, c11;
#pragma unroll
  for (int j = 0; j < 8; ++j) { c00[j] = 0.f; c01[j] = 0.f; c10[j] = 0.f; c11[j] = 0.f; }
  for (int k0 = 0; k0 < Kp; k0 += 32) {
    __builtin_prefetch(b0p + k0 + 64, 0, 1);
    v16bf a0 = load_a_frag(a0p + k0, half);
    v16bf a1 = load_a_frag(a1p + k0, half);
    v16bf b0 = load_b_frag(b0p + k0, half);
    v16bf b1 = load_b_frag(b1p + k0, half);
    c00 = __builtin_amdgcn_wmma_f32_16x16x32_bf16(false, a0, false, b0, (short)0, c00, false, false);
    c01 = __builtin_amdgcn_wmma_f32_16x16x32_bf16(false, a0, false, b1, (short)0, c01, false, false);
    c10 = __builtin_amdgcn_wmma_f32_16x16x32_bf16(false, a1, false, b0, (short)0, c10, false, false);
    c11 = __builtin_amdgcn_wmma_f32_16x16x32_bf16(false, a1, false, b1, (short)0, c11, false, false);
  }
  float bs0 = bias[col0 + mr], bs1 = bias[col0 + 16 + mr];
#pragma unroll
  for (int j = 0; j < 8; ++j) {
    long m0 = row0 + half * 8 + j;
    long m1 = row0 + 16 + half * 8 + j;
    float v;
    v = (c00[j] + bs0) * scale; if (relu) v = fmaxf(v, 0.f);
    Out[m0 * Cout + col0 + mr] = f2bf(v);
    v = (c01[j] + bs1) * scale; if (relu) v = fmaxf(v, 0.f);
    Out[m0 * Cout + col0 + 16 + mr] = f2bf(v);
    v = (c10[j] + bs0) * scale; if (relu) v = fmaxf(v, 0.f);
    Out[m1 * Cout + col0 + mr] = f2bf(v);
    v = (c11[j] + bs1) * scale; if (relu) v = fmaxf(v, 0.f);
    Out[m1 * Cout + col0 + 16 + mr] = f2bf(v);
  }
}

// ---------------------------------------------------------------------------
// Fused SA branch: stage grouped rows in LDS (bf16, zero-padded), 3-layer MLP
// with branch-free WMMA fragment loads, max over samples via shfl.
// Row layout: [points(CF) | xyz - center (3) | zeros..KP]   (sa_msg order)
// LDS: shared0 aliases layer-1 input and layer-2 output (disjoint live ranges).
// When CF is a multiple of 8, the grouped-feature gather is done with
// GLOBAL_LOAD_ASYNC_TO_LDS_B128 (memory->LDS DMA, ASYNCcnt), overlapped with
// the scalar staging of the xyz-center columns.
// ---------------------------------------------------------------------------
template <int NS, int CF, int C1, int C2, int C3>
__global__ __launch_bounds__(128)
void sa_branch_kernel(const float* __restrict__ xyz, int Nin,
                      const __bf16* __restrict__ points,
                      const float* __restrict__ newxyz,
                      const int* __restrict__ gidx,
                      const __bf16* __restrict__ W1, const float* __restrict__ b1,
                      const __bf16* __restrict__ W2, const float* __restrict__ b2,
                      const __bf16* __restrict__ W3, const float* __restrict__ b3,
                      __bf16* __restrict__ out, int outStride, int outOff, int Q) {
  constexpr int K1 = CF + 3;
  constexpr int KP = (K1 + 31) & ~31;
  constexpr int SR = (NS > 64) ? 64 : NS;       // samples per LDS pass
  constexpr int HALVES = NS / SR;
  constexpr int MT = SR / 16;
  constexpr int SH0 = (KP > C2) ? KP : C2;      // shared0 row stride
  constexpr bool ASYNC_STAGE = (CF >= 8) && (CF % 8 == 0);
  __shared__ __align__(32) __bf16 shared0[SR][SH0];  // layer1 input / layer2 out
  __shared__ __align__(32) __bf16 bufA[SR][C1];      // layer1 out
  __shared__ int   rows[SR];
  __shared__ float ctr[3];
  __shared__ float mx[C3];

  int b = blockIdx.y, q = blockIdx.x;
  int tid = threadIdx.x;
  int wave = tid >> 5, lane = tid & 31;
  int half = lane >> 4, mr = lane & 15;

  for (int c = tid; c < C3; c += 128) mx[c] = NEG_INF;
  if (tid < 3) ctr[tid] = newxyz[((long)b * Q + q) * 3 + tid];
  __syncthreads();

  for (int h = 0; h < HALVES; ++h) {
    if (tid < SR) rows[tid] = gidx[((long)b * Q + q) * NS + h * SR + tid];
    __syncthreads();

    // stage grouped input rows (bf16, zero padded to SH0)
    if constexpr (ASYNC_STAGE) {
      // grouped features: per-lane async DMA of aligned 16B chunks into LDS
      constexpr int CHUNKS = CF / 8;
      for (int idx = tid; idx < SR * CHUNKS; idx += 128) {
        int s = idx / CHUNKS, c = (idx % CHUNKS) * 8;
        const __bf16* g = points + ((long)b * Nin + rows[s]) * CF + c;
        unsigned la = (unsigned)(size_t)&shared0[s][c];
        asm volatile("global_load_async_to_lds_b128 %0, %1, off"
                     :: "v"(la), "v"(g) : "memory");
      }
      // xyz - center columns + zero padding (overlaps with async DMA)
      constexpr int REM = SH0 - CF;
      for (int idx = tid; idx < SR * REM; idx += 128) {
        int s = idx / REM, k = CF + idx % REM;
        float v = 0.f;
        if (k < K1) v = xyz[((long)b * Nin + rows[s]) * 3 + (k - CF)] - ctr[k - CF];
        shared0[s][k] = f2bf(v);
      }
      asm volatile("s_wait_asynccnt 0x0" ::: "memory");
    } else {
      for (int idx = tid; idx < SR * SH0; idx += 128) {
        int s = idx / SH0, k = idx % SH0;
        int arow = rows[s];
        float v = 0.f;
        if (k < CF)       v = bf2f(points[((long)b * Nin + arow) * CF + k]);
        else if (k < K1)  v = xyz[((long)b * Nin + arow) * 3 + (k - CF)] - ctr[k - CF];
        shared0[s][k] = f2bf(v);
      }
    }
    __syncthreads();

    // ------- layer 1: shared0 -> bufA
    for (int t = wave; t < MT * (C1 / 16); t += 4) {
      int mt = t / (C1 / 16), nt = t % (C1 / 16);
      v8f c8;
#pragma unroll
      for (int j = 0; j < 8; ++j) c8[j] = 0.f;
      const __bf16* ap = &shared0[mt * 16 + mr][0];
      const __bf16* bp = W1 + (long)(nt * 16 + mr) * KP;
      for (int k0 = 0; k0 < KP; k0 += 32) {
        v16bf a = load_a_frag(ap + k0, half);
        v16bf bm = load_b_frag(bp + k0, half);
        c8 = __builtin_amdgcn_wmma_f32_16x16x32_bf16(false, a, false, bm, (short)0, c8, false, false);
      }
      int n = nt * 16 + mr;
      float bs = b1[n];
#pragma unroll
      for (int j = 0; j < 8; ++j)
        bufA[mt * 16 + half * 8 + j][n] = f2bf(fmaxf((c8[j] + bs) * INV_BN, 0.f));
    }
    __syncthreads();

    // ------- layer 2: bufA -> shared0 (as bufB)
    for (int t = wave; t < MT * (C2 / 16); t += 4) {
      int mt = t / (C2 / 16), nt = t % (C2 / 16);
      v8f c8;
#pragma unroll
      for (int j = 0; j < 8; ++j) c8[j] = 0.f;
      const __bf16* ap = &bufA[mt * 16 + mr][0];
      const __bf16* bp = W2 + (long)(nt * 16 + mr) * C1;
      for (int k0 = 0; k0 < C1; k0 += 32) {
        v16bf a = load_a_frag(ap + k0, half);
        v16bf bm = load_b_frag(bp + k0, half);
        c8 = __builtin_amdgcn_wmma_f32_16x16x32_bf16(false, a, false, bm, (short)0, c8, false, false);
      }
      int n = nt * 16 + mr;
      float bs = b2[n];
#pragma unroll
      for (int j = 0; j < 8; ++j)
        shared0[mt * 16 + half * 8 + j][n] = f2bf(fmaxf((c8[j] + bs) * INV_BN, 0.f));
    }
    __syncthreads();

    // ------- layer 3: shared0 -> max over samples (registers + shfl)
    for (int nt = wave; nt < C3 / 16; nt += 4) {
      float acc = NEG_INF;
      int n = nt * 16 + mr;
      float bs = b3[n];
      const __bf16* bp = W3 + (long)n * C2;
      for (int mt = 0; mt < MT; ++mt) {
        v8f c8;
#pragma unroll
        for (int j = 0; j < 8; ++j) c8[j] = 0.f;
        const __bf16* ap = &shared0[mt * 16 + mr][0];
        for (int k0 = 0; k0 < C2; k0 += 32) {
          v16bf a = load_a_frag(ap + k0, half);
          v16bf bm = load_b_frag(bp + k0, half);
          c8 = __builtin_amdgcn_wmma_f32_16x16x32_bf16(false, a, false, bm, (short)0, c8, false, false);
        }
#pragma unroll
        for (int j = 0; j < 8; ++j)
          acc = fmaxf(acc, fmaxf((c8[j] + bs) * INV_BN, 0.f));
      }
      float other = __shfl_xor(acc, 16, 32);   // combine M half-tiles (same column)
      acc = fmaxf(acc, other);
      if (half == 0) mx[n] = fmaxf(mx[n], acc);
    }
    __syncthreads();
  }

  for (int c = tid; c < C3; c += 128)
    out[((long)b * Q + q) * outStride + outOff + c] = f2bf(mx[c]);
}

// ---------------------------------------------------------------------------
// Head layer 2: 128 -> 1 dot product + sigmoid (f32 weights, bf16 features)
// ---------------------------------------------------------------------------
__global__ void head_out_kernel(const __bf16* __restrict__ h, const float* __restrict__ W,
                                const float* __restrict__ bias, float* __restrict__ out, int R) {
  long i = (long)blockIdx.x * blockDim.x + threadIdx.x;
  if (i >= R) return;
  const __bf16* r = h + i * 128;
  float s = bias[0];
  for (int k = 0; k < 128; ++k) s += bf2f(r[k]) * W[k];
  out[i] = 1.f / (1.f + expf(-s));
}

// ---------------------------------------------------------------------------
// Host orchestration
// ---------------------------------------------------------------------------
static inline long cdiv(long a, long b) { return (a + b - 1) / b; }

extern "C" void kernel_launch(void* const* d_in, const int* in_sizes, int n_in,
                              void* d_out, int out_size, void* d_ws, size_t ws_size,
                              hipStream_t stream) {
  (void)in_sizes; (void)out_size; (void)ws_size;
  const int Bn = 4, Nn = 4000;
  const float* x      = (const float*)d_in[0];
  const float* colors = (const float*)d_in[1];
  const float* P[68];
  for (int i = 0; i < 68 && (2 + i) < n_in; ++i) P[i] = (const float*)d_in[2 + i];
  // param bases: sa1=0(18), sa2=18(18), sa3=36(6), sa4=42(6,unused),
  //              fp3=48(4), fp2=52(4), fp1=56(4), head1=60(4), head2=64(4)

  char* wsp = (char*)d_ws;
  auto alloc = [&](size_t bytes) -> void* {
    void* p = (void*)wsp;
    wsp += (bytes + 255) & ~(size_t)255;
    return p;
  };
  auto wcvt = [&](int pidx, int K, int Kp, int Cout) -> __bf16* {
    __bf16* dst = (__bf16*)alloc(sizeof(__bf16) * (size_t)Cout * Kp);
    long n = (long)Cout * Kp;
    wcvt_kernel<<<cdiv(n, 256), 256, 0, stream>>>(P[pidx], K, Kp, Cout, dst);
    return dst;
  };

  float*  xyz0  = (float*)alloc(sizeof(float) * Bn * Nn * 3);
  __bf16* feat0 = (__bf16*)alloc(sizeof(__bf16) * Bn * Nn * 3);
  float*  cxyz  = (float*)alloc(sizeof(float) * Bn * Nn * 3);
  int*    fidxC = (int*)alloc(sizeof(int) * Bn * 4000);
  float*  cnew  = (float*)alloc(sizeof(float) * Bn * 4000 * 3);
  int*    fidx1 = (int*)alloc(sizeof(int) * Bn * 1000);
  float*  nx1   = (float*)alloc(sizeof(float) * Bn * 1000 * 3);
  int*    fidx2 = (int*)alloc(sizeof(int) * Bn * 512);
  float*  nx2   = (float*)alloc(sizeof(float) * Bn * 512 * 3);
  int*    gidx  = (int*)alloc(sizeof(int) * Bn * 1000 * 128);   // reused per branch
  __bf16* l1    = (__bf16*)alloc(sizeof(__bf16) * Bn * 1000 * 320);
  __bf16* l2    = (__bf16*)alloc(sizeof(__bf16) * Bn * 512 * 640);
  __bf16* sa3in = (__bf16*)alloc(sizeof(__bf16) * Bn * 512 * 672);
  __bf16* t1    = (__bf16*)alloc(sizeof(__bf16) * Bn * 512 * 256);
  __bf16* t2    = (__bf16*)alloc(sizeof(__bf16) * Bn * 512 * 512);
  __bf16* t3    = (__bf16*)alloc(sizeof(__bf16) * Bn * 512 * 1024);
  float*  l3    = (float*)alloc(sizeof(float) * Bn * 1024);
  __bf16* fp3in = (__bf16*)alloc(sizeof(__bf16) * Bn * 512 * 1664);
  __bf16* fp3t  = (__bf16*)alloc(sizeof(__bf16) * Bn * 512 * 256);
  __bf16* l2p   = (__bf16*)alloc(sizeof(__bf16) * Bn * 512 * 256);
  __bf16* fp2in = (__bf16*)alloc(sizeof(__bf16) * Bn * 1000 * 576);
  __bf16* fp2t  = (__bf16*)alloc(sizeof(__bf16) * Bn * 1000 * 256);
  __bf16* l1p   = (__bf16*)alloc(sizeof(__bf16) * Bn * 1000 * 128);
  __bf16* fp1in = (__bf16*)alloc(sizeof(__bf16) * Bn * 4000 * 160);
  __bf16* fp1t  = (__bf16*)alloc(sizeof(__bf16) * Bn * 4000 * 128);
  __bf16* l0p   = (__bf16*)alloc(sizeof(__bf16) * Bn * 4000 * 128);
  __bf16* featb = (__bf16*)alloc(sizeof(__bf16) * Bn * 4000 * 160);
  __bf16* hbuf  = (__bf16*)alloc(sizeof(__bf16) * Bn * 4000 * 128);
  float*  outp  = (float*)d_out;

  // --- pre-convert weights to bf16, K padded to multiple of 32 ---
  __bf16 *sa1w[9], *sa2w[9], *sa3w[3], *fp3w[2], *fp2w[2], *fp1w[2], *h1w, *h2w;
  {
    const int k1[3][3]  = {{6,32,32},{6,64,64},{6,64,96}};
    const int c1[3][3]  = {{32,32,64},{64,64,128},{64,96,128}};
    for (int br = 0; br < 3; ++br)
      for (int l = 0; l < 3; ++l)
        sa1w[br*3+l] = wcvt(br*6 + l*2, k1[br][l], (k1[br][l]+31)&~31, c1[br][l]);
    const int k2[3][3]  = {{323,64,64},{323,128,128},{323,128,128}};
    const int c2[3][3]  = {{64,64,128},{128,128,256},{128,128,256}};
    for (int br = 0; br < 3; ++br)
      for (int l = 0; l < 3; ++l)
        sa2w[br*3+l] = wcvt(18 + br*6 + l*2, k2[br][l], (k2[br][l]+31)&~31, c2[br][l]);
    sa3w[0] = wcvt(36, 643, 672, 256);
    sa3w[1] = wcvt(38, 256, 256, 512);
    sa3w[2] = wcvt(40, 512, 512, 1024);
    fp3w[0] = wcvt(48, 1664, 1664, 256);
    fp3w[1] = wcvt(50, 256, 256, 256);
    fp2w[0] = wcvt(52, 576, 576, 256);
    fp2w[1] = wcvt(54, 256, 256, 128);
    fp1w[0] = wcvt(56, 134, 160, 128);
    fp1w[1] = wcvt(58, 128, 128, 128);
    h1w     = wcvt(60, 131, 160, 128);
    h2w     = wcvt(64, 131, 160, 128);
  }

  // --- split / transpose inputs ---
  {
    long n = 4L * Nn * 3;
    transpose_cn_kernel<<<cdiv(n, 256), 256, 0, stream>>>(x, 6, Nn, 0, 3, xyz0);
    transpose_cn_bf_kernel<<<cdiv(n, 256), 256, 0, stream>>>(x, 6, Nn, 3, 3, feat0);
    transpose_cn_kernel<<<cdiv(n, 256), 256, 0, stream>>>(colors, 3, Nn, 0, 3, cxyz);
  }

  // --- color path (sa4 MLP output unused: only FPS + gather) ---
  fps_kernel<<<Bn, 256, 0, stream>>>(cxyz, 4000, 4000, fidxC);
  gather_rows_kernel<<<cdiv(4L * 4000 * 3, 256), 256, 0, stream>>>(cxyz, 4000, fidxC, 4000, cnew);

  // --- sa1 (MSG) ---
  fps_kernel<<<Bn, 256, 0, stream>>>(xyz0, 4000, 1000, fidx1);
  gather_rows_kernel<<<cdiv(4L * 1000 * 3, 256), 256, 0, stream>>>(xyz0, 4000, fidx1, 1000, nx1);
  {
    dim3 g(1000, Bn);
    query_ball_kernel<<<1000, 128, 0, stream>>>(xyz0, 4000, nx1, 1000, 0.001f * 0.001f, 32, gidx);
    sa_branch_kernel<32, 3, 32, 32, 64><<<g, 128, 0, stream>>>(
        xyz0, 4000, feat0, nx1, gidx,
        sa1w[0], P[1], sa1w[1], P[3], sa1w[2], P[5], l1, 320, 0, 1000);
    query_ball_kernel<<<1000, 128, 0, stream>>>(xyz0, 4000, nx1, 1000, 0.002f * 0.002f, 64, gidx);
    sa_branch_kernel<64, 3, 64, 64, 128><<<g, 128, 0, stream>>>(
        xyz0, 4000, feat0, nx1, gidx,
        sa1w[3], P[7], sa1w[4], P[9], sa1w[5], P[11], l1, 320, 64, 1000);
    query_ball_kernel<<<1000, 128, 0, stream>>>(xyz0, 4000, nx1, 1000, 0.004f * 0.004f, 128, gidx);
    sa_branch_kernel<128, 3, 64, 96, 128><<<g, 128, 0, stream>>>(
        xyz0, 4000, feat0, nx1, gidx,
        sa1w[6], P[13], sa1w[7], P[15], sa1w[8], P[17], l1, 320, 192, 1000);
  }

  // --- sa2 (MSG) ---
  fps_kernel<<<Bn, 256, 0, stream>>>(nx1, 1000, 512, fidx2);
  gather_rows_kernel<<<cdiv(4L * 512 * 3, 256), 256, 0, stream>>>(nx1, 1000, fidx2, 512, nx2);
  {
    dim3 g(512, Bn);
    query_ball_kernel<<<512, 128, 0, stream>>>(nx1, 1000, nx2, 512, 0.004f * 0.004f, 64, gidx);
    sa_branch_kernel<64, 320, 64, 64, 128><<<g, 128, 0, stream>>>(
        nx1, 1000, l1, nx2, gidx,
        sa2w[0], P[19], sa2w[1], P[21], sa2w[2], P[23], l2, 640, 0, 512);
    query_ball_kernel<<<512, 128, 0, stream>>>(nx1, 1000, nx2, 512, 0.006f * 0.006f, 64, gidx);
    sa_branch_kernel<64, 320, 128, 128, 256><<<g, 128, 0, stream>>>(
        nx1, 1000, l1, nx2, gidx,
        sa2w[3], P[25], sa2w[4], P[27], sa2w[5], P[29], l2, 640, 128, 512);
    query_ball_kernel<<<512, 128, 0, stream>>>(nx1, 1000, nx2, 512, 0.008f * 0.008f, 128, gidx);
    sa_branch_kernel<128, 320, 128, 128, 256><<<g, 128, 0, stream>>>(
        nx1, 1000, l1, nx2, gidx,
        sa2w[6], P[31], sa2w[7], P[33], sa2w[8], P[35], l2, 640, 384, 512);
  }

  // --- sa3 (group all): concat [xyz | points] = 643 (pad 672), MLP, max ---
  {
    long rows = 4L * 512;
    copy_f32_bf_cols_kernel<<<cdiv(rows * 3, 256), 256, 0, stream>>>(nx2, 3, sa3in, 672, 0, rows);
    copy_bf_cols_kernel<<<cdiv(rows * 640, 256), 256, 0, stream>>>(l2, 640, sa3in, 672, 3, rows);
    zero_cols_kernel<<<cdiv(rows * (672 - 643), 256), 256, 0, stream>>>(sa3in, 672, 643, rows);
    wmma_gemm_kernel<<<dim3(rows / 32, 256 / 32), 32, 0, stream>>>(sa3in, sa3w[0], P[37], t1, 672, 256, INV_BN, 1);
    wmma_gemm_kernel<<<dim3(rows / 32, 512 / 32), 32, 0, stream>>>(t1, sa3w[1], P[39], t2, 256, 512, INV_BN, 1);
    wmma_gemm_kernel<<<dim3(rows / 32, 1024 / 32), 32, 0, stream>>>(t2, sa3w[2], P[41], t3, 512, 1024, INV_BN, 1);
    max_rows_kernel<<<cdiv(4L * 1024, 256), 256, 0, stream>>>(t3, 512, 1024, l3);
  }

  // --- fp3 (S==1): concat [l2(640) | broadcast l3(1024)] = 1664 ---
  {
    long rows = 4L * 512;
    copy_bf_cols_kernel<<<cdiv(rows * 640, 256), 256, 0, stream>>>(l2, 640, fp3in, 1664, 0, rows);
    broadcast_rows_kernel<<<cdiv(4L * 512 * 1024, 256), 256, 0, stream>>>(l3, 1024, fp3in, 1664, 640, 512);
    wmma_gemm_kernel<<<dim3(rows / 32, 256 / 32), 32, 0, stream>>>(fp3in, fp3w[0], P[49], fp3t, 1664, 256, INV_BN, 1);
    wmma_gemm_kernel<<<dim3(rows / 32, 256 / 32), 32, 0, stream>>>(fp3t, fp3w[1], P[51], l2p, 256, 256, INV_BN, 1);
  }

  // --- fp2: concat [l1(320) | 3NN interp l2p(256)] = 576 ---
  {
    long rows = 4L * 1000;
    copy_bf_cols_kernel<<<cdiv(rows * 320, 256), 256, 0, stream>>>(l1, 320, fp2in, 576, 0, rows);
    knn3_interp_kernel<<<cdiv(rows, 128), 128, 0, stream>>>(nx1, 1000, nx2, 512, l2p, 256, fp2in, 576, 320);
    wmma_gemm_kernel<<<dim3(rows / 32, 256 / 32), 32, 0, stream>>>(fp2in, fp2w[0], P[53], fp2t, 576, 256, INV_BN, 1);
    wmma_gemm_kernel<<<dim3(rows / 32, 128 / 32), 32, 0, stream>>>(fp2t, fp2w[1], P[55], l1p, 256, 128, INV_BN, 1);
  }

  // --- fp1: concat [xyz(3) | feat(3) | 3NN interp l1p(128)] = 134 (pad 160) ---
  {
    long rows = 4L * 4000;
    copy_f32_bf_cols_kernel<<<cdiv(rows * 3, 256), 256, 0, stream>>>(xyz0, 3, fp1in, 160, 0, rows);
    copy_bf_cols_kernel<<<cdiv(rows * 3, 256), 256, 0, stream>>>(feat0, 3, fp1in, 160, 3, rows);
    knn3_interp_kernel<<<cdiv(rows, 128), 128, 0, stream>>>(xyz0, 4000, nx1, 1000, l1p, 128, fp1in, 160, 6);
    zero_cols_kernel<<<cdiv(rows * (160 - 134), 256), 256, 0, stream>>>(fp1in, 160, 134, rows);
    wmma_gemm_kernel<<<dim3(rows / 32, 128 / 32), 32, 0, stream>>>(fp1in, fp1w[0], P[57], fp1t, 160, 128, INV_BN, 1);
    wmma_gemm_kernel<<<dim3(rows / 32, 128 / 32), 32, 0, stream>>>(fp1t, fp1w[1], P[59], l0p, 128, 128, INV_BN, 1);
  }

  // --- heads: feat = [l0p(128) | color_new_xyz(3)] = 131 (pad 160) ---
  {
    long rows = 4L * 4000;
    copy_bf_cols_kernel<<<cdiv(rows * 128, 256), 256, 0, stream>>>(l0p, 128, featb, 160, 0, rows);
    copy_f32_bf_cols_kernel<<<cdiv(rows * 3, 256), 256, 0, stream>>>(cnew, 3, featb, 160, 128, rows);
    zero_cols_kernel<<<cdiv(rows * (160 - 131), 256), 256, 0, stream>>>(featb, 160, 131, rows);
    // head1: linear (no relu) * INV_BN, then 128->1 sigmoid
    wmma_gemm_kernel<<<dim3(rows / 32, 128 / 32), 32, 0, stream>>>(featb, h1w, P[61], hbuf, 160, 128, INV_BN, 0);
    head_out_kernel<<<cdiv(rows, 256), 256, 0, stream>>>(hbuf, P[62], P[63], outp, (int)rows);
    // head2
    wmma_gemm_kernel<<<dim3(rows / 32, 128 / 32), 32, 0, stream>>>(featb, h2w, P[65], hbuf, 160, 128, INV_BN, 0);
    head_out_kernel<<<cdiv(rows, 256), 256, 0, stream>>>(hbuf, P[66], P[67], outp + rows, (int)rows);
  }
}
